// FrameAggregator_56435870269645
// MI455X (gfx1250) — matrix operand order repaired and checked by reading, not compiled, not run on hardware
//
#include <hip/hip_runtime.h>
#include <math.h>

// ---------------- problem constants ----------------
#define B_      128
#define N_      512
#define E_      (128 * 8192)   // 1048576 edges
#define NT_     (B_ * N_)      // 65536 nodes
#define IN_CH_  128
#define HID_    64
#define K1_     64             // 2*K clusters, pool 1
#define K2_     32             // K clusters, pool 2
#define OUT_CH_ 128
#define EPS_    1e-15f

typedef __attribute__((ext_vector_type(16))) _Float16 v16h;
typedef __attribute__((ext_vector_type(8)))  float    v8f;
typedef unsigned int u32x4 __attribute__((ext_vector_type(4)));
typedef int          i32x4 __attribute__((ext_vector_type(4)));
typedef int          i32x8 __attribute__((ext_vector_type(8)));

#if __has_builtin(__builtin_amdgcn_tensor_load_to_lds)
#define HAS_TDM 1
#else
#define HAS_TDM 0
#endif
#if __has_builtin(__builtin_amdgcn_s_wait_tensorcnt)
#define HAS_TCNT 1
#else
#define HAS_TCNT 0
#endif

// ---------------- WMMA helpers (gfx1250 wave32 layouts, ISA 7.12.2) ------

__device__ __forceinline__ v8f wmma_f16(v16h a, v16h b, v8f c) {
  return __builtin_amdgcn_wmma_f32_16x16x32_f16(false, a, false, b, (short)0, c,
                                                false, false);
}

// A fragment 16x32 f16 from row-major f32; lane L row m=L%16;
// VGPR v pair (k,k+1): k = (v%4)*2 + (v/4)*16 + (L/16)*8
__device__ __forceinline__ v16h load_a(const float* __restrict__ A, int lda,
                                       int row0, int k0, int lane) {
  int m = lane & 15, koff = (lane >> 4) * 8;
  const float* base = A + (size_t)(row0 + m) * lda + k0 + koff;
  v16h a;
#pragma unroll
  for (int v = 0; v < 8; ++v) {
    int k = (v & 3) * 2 + (v >> 2) * 16;
    a[2 * v]     = (_Float16)base[k];
    a[2 * v + 1] = (_Float16)base[k + 1];
  }
  return a;
}

// same but from an f16 source (pairs contiguous -> b32 loads)
__device__ __forceinline__ v16h load_a_h(const _Float16* __restrict__ A, int lda,
                                         int row0, int k0, int lane) {
  int m = lane & 15, koff = (lane >> 4) * 8;
  const _Float16* base = A + (size_t)(row0 + m) * lda + k0 + koff;
  v16h a;
#pragma unroll
  for (int v = 0; v < 8; ++v) {
    int k = (v & 3) * 2 + (v >> 2) * 16;
    a[2 * v]     = base[k];
    a[2 * v + 1] = base[k + 1];
  }
  return a;
}

// A = S^T, S row-major f16: element (m,k) = S[k][m0+m]
__device__ __forceinline__ v16h load_at_h(const _Float16* __restrict__ S, int ldS,
                                          int m0, int k0, int lane) {
  int m = lane & 15, koff = (lane >> 4) * 8;
  v16h a;
#pragma unroll
  for (int v = 0; v < 8; ++v) {
    int k = k0 + koff + (v & 3) * 2 + (v >> 2) * 16;
    a[2 * v]     = S[(size_t)k * ldS + m0 + m];
    a[2 * v + 1] = S[(size_t)(k + 1) * ldS + m0 + m];
  }
  return a;
}

__device__ __forceinline__ v16h load_at(const float* __restrict__ S, int ldS,
                                        int m0, int k0, int lane) {
  int m = lane & 15, koff = (lane >> 4) * 8;
  v16h a;
#pragma unroll
  for (int v = 0; v < 8; ++v) {
    int k = k0 + koff + (v & 3) * 2 + (v >> 2) * 16;
    a[2 * v]     = (_Float16)S[(size_t)k * ldS + m0 + m];
    a[2 * v + 1] = (_Float16)S[(size_t)(k + 1) * ldS + m0 + m];
  }
  return a;
}

// B fragment 32x16 from row-major KxN f32: lane L col n=L%16; k=(v%4)*2+(v/4)*16+(L/16)*8
__device__ __forceinline__ v16h load_b(const float* __restrict__ Bm, int ldb,
                                       int k0, int col0, int lane) {
  int n = lane & 15, kh = (lane >> 4) * 8;
  v16h b;
#pragma unroll
  for (int v = 0; v < 8; ++v) {
    int k = k0 + kh + (v & 3) * 2 + (v >> 2) * 16;
    b[2 * v]     = (_Float16)Bm[(size_t)k * ldb + col0 + n];
    b[2 * v + 1] = (_Float16)Bm[(size_t)(k + 1) * ldb + col0 + n];
  }
  return b;
}

__device__ __forceinline__ v16h load_b_h(const _Float16* __restrict__ Bm, int ldb,
                                         int k0, int col0, int lane) {
  int n = lane & 15, kh = (lane >> 4) * 8;
  v16h b;
#pragma unroll
  for (int v = 0; v < 8; ++v) {
    int k = k0 + kh + (v & 3) * 2 + (v >> 2) * 16;
    b[2 * v]     = Bm[(size_t)k * ldb + col0 + n];
    b[2 * v + 1] = Bm[(size_t)(k + 1) * ldb + col0 + n];
  }
  return b;
}

// C/D 16x16 f32: VGPR v, lane L -> row m = v + 8*(L/16), col n = L%16
__device__ __forceinline__ void store_c(float* __restrict__ C, int ldc, int row0,
                                        int col0, int lane, v8f c) {
  int n = lane & 15, mh = (lane >> 4) * 8;
#pragma unroll
  for (int v = 0; v < 8; ++v)
    C[(size_t)(row0 + mh + v) * ldc + col0 + n] = c[v];
}

__device__ __forceinline__ void store_c_bias(float* __restrict__ C, int ldc,
                                             int row0, int col0, int lane, v8f c,
                                             const float* __restrict__ bias,
                                             int do_relu) {
  int n = lane & 15, mh = (lane >> 4) * 8;
  float bv = bias[col0 + n];
#pragma unroll
  for (int v = 0; v < 8; ++v) {
    float x = c[v] + bv;
    if (do_relu) x = fmaxf(x, 0.f);
    C[(size_t)(row0 + mh + v) * ldc + col0 + n] = x;
  }
}

__device__ __forceinline__ float block_reduce256(float v, float* red) {
  int tid = threadIdx.x;
  red[tid] = v;
  __syncthreads();
#pragma unroll
  for (int s = 128; s > 0; s >>= 1) {
    if (tid < s) red[tid] += red[tid + s];
    __syncthreads();
  }
  float r = red[0];
  __syncthreads();
  return r;
}

// ---------------- Tensor Data Mover slab loader ----------------
// Loads a 32-row x 64-col f16 tile (row stride 64) from global into LDS.
// D# per ISA 8.3/8.4: group0 = {count/type, lds_addr, global_addr},
// group1 = {data_size, tensor dims/strides, tile dims}. 2D -> groups 2/3 zero.
__device__ __forceinline__ void tdm_load_slab(const _Float16* gsrc,
                                              _Float16* ldst, int rows_total) {
#if HAS_TDM
  unsigned long long ga = (unsigned long long)(size_t)gsrc;
  unsigned lds_addr = (unsigned)(size_t)ldst;  // LDS aperture: addr[31:0]
  u32x4 g0;
  g0[0] = 1u;                                   // count=1, user mode
  g0[1] = lds_addr;
  g0[2] = (unsigned)(ga & 0xffffffffu);
  g0[3] = (unsigned)((ga >> 32) & 0x01ffffffu) | (2u << 30);  // type=2 (image)
  i32x8 g1;
  g1[0] = 1 << 16;            // data_size=1 (2 bytes), no multicast/pad
  g1[1] = 64 << 16;           // tensor_dim0 = 64 (lo16 at bits 63:48)
  g1[2] = rows_total << 16;   // dim0 hi=0 | tensor_dim1 (rows) lo16
  g1[3] = 64 << 16;           // dim1 hi=0 | tile_dim0 = 64
  g1[4] = 32;                 // tile_dim1 = 32 rows, tile_dim2 = 0
  g1[5] = 64;                 // tensor_dim0_stride = 64 elements
  g1[6] = 0;
  g1[7] = 0;
  i32x4 z4 = {};
#if defined(__clang_major__) && (__clang_major__ >= 23)
  i32x8 z8 = {};
  __builtin_amdgcn_tensor_load_to_lds(g0, g1, z4, z4, z8, 0);
#else
  __builtin_amdgcn_tensor_load_to_lds(g0, g1, z4, z4, 0);
#endif
#else
  // fallback: wave-0 synchronous copy
  int lane = threadIdx.x & 31;
  for (int i = lane; i < 32 * 64; i += 32) ldst[i] = gsrc[i];
  (void)rows_total;
#endif
}

__device__ __forceinline__ void tdm_wait() {
#if HAS_TDM && HAS_TCNT
  __builtin_amdgcn_s_wait_tensorcnt(0);
#endif
}

// ---------------- elementwise / scatter kernels ----------------

__global__ void k_fill(float* p, float v, int n) {
  int i = blockIdx.x * blockDim.x + threadIdx.x;
  if (i < n) p[i] = v;
}

__global__ void k_deg_scatter(const int* __restrict__ col,
                              const float* __restrict__ ew, float* deg) {
  int e = blockIdx.x * 256 + threadIdx.x;
  if (e >= E_) return;
  atomicAdd(&deg[col[e]], ew[e]);
}

__global__ void k_dis(float* dis) {
  int i = blockIdx.x * 256 + threadIdx.x;
  if (i >= NT_) return;
  float d = dis[i];
  dis[i] = (d > 0.f) ? rsqrtf(fmaxf(d, 1e-12f)) : 0.f;
}

__global__ void k_h_init(const float* __restrict__ dis,
                         const float* __restrict__ xw,
                         const float* __restrict__ b1, float* h) {
  int idx = blockIdx.x * 256 + threadIdx.x;
  if (idx >= NT_ * HID_) return;
  int i = idx >> 6, f = idx & 63;
  float dv = dis[i];
  h[idx] = b1[f] + dv * dv * xw[idx];
}

__global__ void k_h_scatter(const int* __restrict__ row,
                            const int* __restrict__ col,
                            const float* __restrict__ ew,
                            const float* __restrict__ dis,
                            const float* __restrict__ xw, float* h) {
  long long idx = (long long)blockIdx.x * 256 + threadIdx.x;
  if (idx >= (long long)E_ * HID_) return;
  int e = (int)(idx >> 6), f = (int)(idx & 63);
  int r = row[e], c = col[e];
  float nrm = dis[r] * ew[e] * dis[c];
  atomicAdd(&h[(size_t)c * HID_ + f], nrm * xw[(size_t)r * HID_ + f]);
}

// relu in place + fp16 mirror for the WMMA-consuming stages
__global__ void k_relu_cvt(float* p, _Float16* ph, int n) {
  int i = blockIdx.x * 256 + threadIdx.x;
  if (i >= n) return;
  float v = fmaxf(p[i], 0.f);
  p[i] = v;
  ph[i] = (_Float16)v;
}

__global__ void k_cvt_h(const float* __restrict__ src, _Float16* dst, int n) {
  int i = blockIdx.x * 256 + threadIdx.x;
  if (i < n) dst[i] = (_Float16)src[i];
}

__global__ void k_As_scatter(const int* __restrict__ row,
                             const int* __restrict__ col,
                             const float* __restrict__ s1, float* As1,
                             float* dflat1) {
  long long idx = (long long)blockIdx.x * 256 + threadIdx.x;
  if (idx >= (long long)E_ * K1_) return;
  int e = (int)(idx >> 6), f = (int)(idx & 63);
  int r = row[e], c = col[e];
  atomicAdd(&As1[(size_t)r * K1_ + f], s1[(size_t)c * K1_ + f]);
  if (f == 0) atomicAdd(&dflat1[r], 1.0f);
}

// ---------------- WMMA GEMM kernels ----------------

__global__ void k_gemm_xw(const float* __restrict__ pos,
                          const float* __restrict__ W1, float* xw) {
  int lane = threadIdx.x & 31, w = threadIdx.x >> 5;
  int row0 = blockIdx.x * 32 + (w >> 2) * 16;
  int col0 = (w & 3) * 16;
  v8f acc = {};
#pragma unroll
  for (int k = 0; k < IN_CH_; k += 32)
    acc = wmma_f16(load_a(pos, IN_CH_, row0, k, lane),
                   load_b(W1, HID_, k, col0, lane), acc);
  store_c(xw, HID_, row0, col0, lane, acc);
}

__global__ void k_gemm_s1(const float* __restrict__ x1,
                          const float* __restrict__ Wp1,
                          const float* __restrict__ bp1, float* s1) {
  int lane = threadIdx.x & 31, w = threadIdx.x >> 5;
  int row0 = blockIdx.x * 32 + (w >> 2) * 16;
  int col0 = (w & 3) * 16;
  v8f acc = {};
#pragma unroll
  for (int k = 0; k < HID_; k += 32)
    acc = wmma_f16(load_a(x1, HID_, row0, k, lane),
                   load_b(Wp1, K1_, k, col0, lane), acc);
  store_c_bias(s1, K1_, row0, col0, lane, acc, bp1, 0);
}

__global__ void k_softmax64(float* s, _Float16* sh) {  // wave per row of 64
  int gt = blockIdx.x * blockDim.x + threadIdx.x;
  int wid = gt >> 5, lane = threadIdx.x & 31;
  if (wid >= NT_) return;
  float* p = s + (size_t)wid * 64;
  _Float16* ph = sh + (size_t)wid * 64;
  float a = p[lane], b = p[lane + 32];
  float m = fmaxf(a, b);
#pragma unroll
  for (int o = 16; o > 0; o >>= 1) m = fmaxf(m, __shfl_xor(m, o, 32));
  a = __expf(a - m);
  b = __expf(b - m);
  float sum = a + b;
#pragma unroll
  for (int o = 16; o > 0; o >>= 1) sum += __shfl_xor(sum, o, 32);
  float inv = 1.f / sum;
  a *= inv;
  b *= inv;
  p[lane] = a;
  p[lane + 32] = b;
  ph[lane] = (_Float16)a;
  ph[lane + 32] = (_Float16)b;
}

__global__ void k_softmax32(float* s, int rows) {
  int gt = blockIdx.x * blockDim.x + threadIdx.x;
  int wid = gt >> 5, lane = threadIdx.x & 31;
  if (wid >= rows) return;
  float* p = s + (size_t)wid * 32;
  float a = p[lane];
  float m = a;
#pragma unroll
  for (int o = 16; o > 0; o >>= 1) m = fmaxf(m, __shfl_xor(m, o, 32));
  a = __expf(a - m);
  float sum = a;
#pragma unroll
  for (int o = 16; o > 0; o >>= 1) sum += __shfl_xor(sum, o, 32);
  p[lane] = a / sum;
}

// pool1 per batch: out1 = s1^T x1, oadj1 = s1^T As1, ss1 = s1^T s1. K = 512.
// K-slabs (32x64 f16) of all three operands TDM-loaded into LDS, double
// buffered so the DMA overlaps the WMMA stream. 8 waves x 2 tiles x 3 accs.
__global__ void k_pool1(const _Float16* __restrict__ s1h,
                        const _Float16* __restrict__ x1h,
                        const _Float16* __restrict__ As1h, float* out1,
                        float* oadj1, float* ss1) {
  __shared__ _Float16 sS[2][32 * 64];
  __shared__ _Float16 sX[2][32 * 64];
  __shared__ _Float16 sA[2][32 * 64];
  int b = blockIdx.x;
  int lane = threadIdx.x & 31, w = threadIdx.x >> 5;
  const _Float16* sb = s1h + (size_t)b * N_ * K1_;
  const _Float16* xb = x1h + (size_t)b * N_ * HID_;
  const _Float16* ab = As1h + (size_t)b * N_ * K1_;

  int mt0 = (w >> 2) * 16;        // tiles w and w+8
  int mt1 = ((w + 8) >> 2) * 16;
  int nt = (w & 3) * 16;
  v8f c00 = {}, c01 = {}, c02 = {}, c10 = {}, c11 = {}, c12 = {};

  if (w == 0) {  // prologue: slab 0
    tdm_load_slab(sb, &sS[0][0], N_);
    tdm_load_slab(xb, &sX[0][0], N_);
    tdm_load_slab(ab, &sA[0][0], N_);
    tdm_wait();
  }
  __syncthreads();

  for (int step = 0; step < 16; ++step) {
    int cur = step & 1;
    if (step + 1 < 16 && w == 0) {  // async: next slab while computing
      int k = (step + 1) * 32;
      tdm_load_slab(sb + (size_t)k * 64, &sS[cur ^ 1][0], N_);
      tdm_load_slab(xb + (size_t)k * 64, &sX[cur ^ 1][0], N_);
      tdm_load_slab(ab + (size_t)k * 64, &sA[cur ^ 1][0], N_);
    }
    const _Float16* pS = &sS[cur][0];
    const _Float16* pX = &sX[cur][0];
    const _Float16* pA = &sA[cur][0];
    v16h a0 = load_at_h(pS, 64, mt0, 0, lane);
    v16h a1 = load_at_h(pS, 64, mt1, 0, lane);
    v16h bx = load_b_h(pX, 64, 0, nt, lane);
    v16h ba = load_b_h(pA, 64, 0, nt, lane);
    v16h bs = load_b_h(pS, 64, 0, nt, lane);
    c00 = wmma_f16(a0, bx, c00);
    c01 = wmma_f16(a0, ba, c01);
    c02 = wmma_f16(a0, bs, c02);
    c10 = wmma_f16(a1, bx, c10);
    c11 = wmma_f16(a1, ba, c11);
    c12 = wmma_f16(a1, bs, c12);
    if (step + 1 < 16) {
      if (w == 0) tdm_wait();
      __syncthreads();
    }
  }
  float* o1 = out1 + (size_t)b * 4096;
  float* oa = oadj1 + (size_t)b * 4096;
  float* os = ss1 + (size_t)b * 4096;
  store_c(o1, 64, mt0, nt, lane, c00);
  store_c(oa, 64, mt0, nt, lane, c01);
  store_c(os, 64, mt0, nt, lane, c02);
  store_c(o1, 64, mt1, nt, lane, c10);
  store_c(oa, 64, mt1, nt, lane, c11);
  store_c(os, 64, mt1, nt, lane, c12);
}

// stats for pool 1: losses + symmetric normalization -> adj1n, dflat2
__global__ void k_stats1(const float* __restrict__ s1,
                         const float* __restrict__ dflat1,
                         const float* __restrict__ oadj1,
                         const float* __restrict__ ss1, float* adj1n,
                         float* dflat2, float* mc_out, float* o_out) {
  __shared__ float red[256];
  __shared__ float dsh[64];
  int b = blockIdx.x, tid = threadIdx.x;
  const float* sb = s1 + (size_t)b * N_ * K1_;
  const float* oa = oadj1 + (size_t)b * 4096;
  const float* ssb = ss1 + (size_t)b * 4096;

  float den = 0.f;
  for (int i = tid; i < N_ * K1_; i += 256) {
    float v = sb[i];
    den += v * v * dflat1[b * N_ + (i >> 6)];
  }
  den = block_reduce256(den, red);
  float tr = block_reduce256(tid < 64 ? oa[tid * 65] : 0.f, red);
  float f2 = 0.f;
  for (int i = tid; i < 4096; i += 256) { float v = ssb[i]; f2 += v * v; }
  f2 = block_reduce256(f2, red);
  float inv_fro = rsqrtf(fmaxf(f2, 1e-30f));
  float is = rsqrtf(64.f);
  float res = 0.f;
  for (int i = tid; i < 4096; i += 256) {
    int r = i >> 6, c = i & 63;
    float v = ssb[i] * inv_fro - ((r == c) ? is : 0.f);
    res += v * v;
  }
  float ortho = sqrtf(block_reduce256(res, red));
  if (tid == 0) {
    atomicAdd(mc_out, -(tr / den) * (1.0f / B_));
    atomicAdd(o_out, ortho * (1.0f / B_));
  }
  if (tid < 64) {
    float rs = 0.f;
    for (int j = 0; j < 64; ++j) rs += (j == tid) ? 0.f : oa[tid * 64 + j];
    dsh[tid] = sqrtf(rs) + EPS_;
  }
  __syncthreads();
  if (tid < 64) {
    float di = dsh[tid], rsum = 0.f;
    float* rowp = adj1n + (size_t)b * 4096 + tid * 64;
    for (int j = 0; j < 64; ++j) {
      float v = (j == tid) ? 0.f : oa[tid * 64 + j] / (di * dsh[j]);
      rowp[j] = v;
      rsum += v;
    }
    dflat2[b * 64 + tid] = rsum;
  }
}

// t1 = adj1n @ out1 per batch (64x64x64)
__global__ void k_t1(const float* __restrict__ adj1n,
                     const float* __restrict__ out1, float* t1) {
  int b = blockIdx.x, lane = threadIdx.x & 31, w = threadIdx.x >> 5;
  const float* A = adj1n + (size_t)b * 4096;
  const float* Bm = out1 + (size_t)b * 4096;
  for (int t = w; t < 16; t += 8) {
    int mt = (t >> 2) * 16, nt = (t & 3) * 16;
    v8f c = {};
#pragma unroll
    for (int k = 0; k < 64; k += 32)
      c = wmma_f16(load_a(A, 64, mt, k, lane), load_b(Bm, 64, k, nt, lane), c);
    store_c(t1 + (size_t)b * 4096, 64, mt, nt, lane, c);
  }
}

// x2 = relu(t1@W2r + out1@W2s + b2), flat M=8192, K=64, N=64
__global__ void k_x2(const float* __restrict__ t1, const float* __restrict__ out1,
                     const float* __restrict__ W2r, const float* __restrict__ W2s,
                     const float* __restrict__ b2, float* x2) {
  int lane = threadIdx.x & 31, w = threadIdx.x >> 5;
  int tile = blockIdx.x * 8 + w;
  int row0 = (tile >> 2) * 16, col0 = (tile & 3) * 16;
  v8f c = {};
#pragma unroll
  for (int k = 0; k < 64; k += 32) {
    c = wmma_f16(load_a(t1, 64, row0, k, lane), load_b(W2r, 64, k, col0, lane), c);
    c = wmma_f16(load_a(out1, 64, row0, k, lane), load_b(W2s, 64, k, col0, lane), c);
  }
  store_c_bias(x2, 64, row0, col0, lane, c, b2, 1);
}

// s2 = x2 @ Wp2 + bp2, flat M=8192, K=64, N=32
__global__ void k_gemm_s2(const float* __restrict__ x2,
                          const float* __restrict__ Wp2,
                          const float* __restrict__ bp2, float* s2) {
  int lane = threadIdx.x & 31, w = threadIdx.x >> 5;
  int tile = blockIdx.x * 8 + w;
  int row0 = (tile >> 1) * 16, col0 = (tile & 1) * 16;
  v8f c = {};
#pragma unroll
  for (int k = 0; k < 64; k += 32)
    c = wmma_f16(load_a(x2, 64, row0, k, lane), load_b(Wp2, 32, k, col0, lane), c);
  store_c_bias(s2, 32, row0, col0, lane, c, bp2, 0);
}

// pool2 per batch: As2 = adj1n@s2 (staged in LDS), out2 = s2^T x2,
// oadj2 = s2^T As2, ss2 = s2^T s2
__global__ void k_pool2(const float* __restrict__ adj1n,
                        const float* __restrict__ s2,
                        const float* __restrict__ x2, float* out2, float* oadj2,
                        float* ss2) {
  __shared__ float sAs[64 * 32];
  int b = blockIdx.x, lane = threadIdx.x & 31, w = threadIdx.x >> 5;
  const float* A1 = adj1n + (size_t)b * 4096;
  const float* s2b = s2 + (size_t)b * 64 * 32;
  const float* x2b = x2 + (size_t)b * 4096;
  {
    int mt = (w >> 1) * 16, nt = (w & 1) * 16;
    v8f c = {};
#pragma unroll
    for (int k = 0; k < 64; k += 32)
      c = wmma_f16(load_a(A1, 64, mt, k, lane), load_b(s2b, 32, k, nt, lane), c);
    store_c(sAs, 32, mt, nt, lane, c);
  }
  __syncthreads();
  {
    int mt = (w >> 2) * 16, nt = (w & 3) * 16;
    v8f c = {};
#pragma unroll
    for (int k = 0; k < 64; k += 32)
      c = wmma_f16(load_at(s2b, 32, mt, k, lane), load_b(x2b, 64, k, nt, lane), c);
    store_c(out2 + (size_t)b * 2048, 64, mt, nt, lane, c);
  }
  {
    int t = w & 3;
    int mt = (t >> 1) * 16, nt = (t & 1) * 16;
    const float* Bm = (w < 4) ? sAs : s2b;
    v8f c = {};
#pragma unroll
    for (int k = 0; k < 64; k += 32)
      c = wmma_f16(load_at(s2b, 32, mt, k, lane), load_b(Bm, 32, k, nt, lane), c);
    store_c((w < 4 ? oadj2 : ss2) + (size_t)b * 1024, 32, mt, nt, lane, c);
  }
}

__global__ void k_stats2(const float* __restrict__ s2,
                         const float* __restrict__ dflat2,
                         const float* __restrict__ oadj2,
                         const float* __restrict__ ss2, float* adj2n,
                         float* mc_out, float* o_out) {
  __shared__ float red[256];
  __shared__ float dsh[32];
  int b = blockIdx.x, tid = threadIdx.x;
  const float* sb = s2 + (size_t)b * 2048;
  const float* oa = oadj2 + (size_t)b * 1024;
  const float* ssb = ss2 + (size_t)b * 1024;
  float den = 0.f;
  for (int i = tid; i < 2048; i += 256) {
    float v = sb[i];
    den += v * v * dflat2[b * 64 + (i >> 5)];
  }
  den = block_reduce256(den, red);
  float tr = block_reduce256(tid < 32 ? oa[tid * 33] : 0.f, red);
  float f2 = 0.f;
  for (int i = tid; i < 1024; i += 256) { float v = ssb[i]; f2 += v * v; }
  f2 = block_reduce256(f2, red);
  float inv_fro = rsqrtf(fmaxf(f2, 1e-30f));
  float is = rsqrtf(32.f);
  float res = 0.f;
  for (int i = tid; i < 1024; i += 256) {
    int r = i >> 5, c = i & 31;
    float v = ssb[i] * inv_fro - ((r == c) ? is : 0.f);
    res += v * v;
  }
  float ortho = sqrtf(block_reduce256(res, red));
  if (tid == 0) {
    atomicAdd(mc_out, -(tr / den) * (1.0f / B_));
    atomicAdd(o_out, ortho * (1.0f / B_));
  }
  if (tid < 32) {
    float rs = 0.f;
    for (int j = 0; j < 32; ++j) rs += (j == tid) ? 0.f : oa[tid * 32 + j];
    dsh[tid] = sqrtf(rs) + EPS_;
  }
  __syncthreads();
  if (tid < 32) {
    float di = dsh[tid];
    float* rowp = adj2n + (size_t)b * 1024 + tid * 32;
    for (int j = 0; j < 32; ++j)
      rowp[j] = (j == tid) ? 0.f : oa[tid * 32 + j] / (di * dsh[j]);
  }
}

// t2 = adj2n @ out2 per batch (32x32 @ 32x64), single K step
__global__ void k_t2(const float* __restrict__ adj2n,
                     const float* __restrict__ out2, float* t2) {
  int b = blockIdx.x, lane = threadIdx.x & 31, w = threadIdx.x >> 5;
  const float* A = adj2n + (size_t)b * 1024;
  const float* Bm = out2 + (size_t)b * 2048;
  int mt = (w >> 2) * 16, nt = (w & 3) * 16;
  v8f c = {};
  c = wmma_f16(load_a(A, 32, mt, 0, lane), load_b(Bm, 64, 0, nt, lane), c);
  store_c(t2 + (size_t)b * 2048, 64, mt, nt, lane, c);
}

// x3 = t2@W3r + out2@W3s + b3 -> d_out x section. flat M=4096, K=64, N=128
__global__ void k_x3(const float* __restrict__ t2, const float* __restrict__ out2,
                     const float* __restrict__ W3r, const float* __restrict__ W3s,
                     const float* __restrict__ b3, float* xo) {
  int lane = threadIdx.x & 31, w = threadIdx.x >> 5;
  int tile = blockIdx.x * 8 + w;
  int row0 = (tile >> 3) * 16, col0 = (tile & 7) * 16;
  v8f c = {};
#pragma unroll
  for (int k = 0; k < 64; k += 32) {
    c = wmma_f16(load_a(t2, 64, row0, k, lane), load_b(W3r, 128, k, col0, lane), c);
    c = wmma_f16(load_a(out2, 64, row0, k, lane), load_b(W3s, 128, k, col0, lane), c);
  }
  store_c_bias(xo, 128, row0, col0, lane, c, b3, 0);
}

// agg = s1s @ s2s per batch: 512x64 @ 64x32 -> d_out agg section
__global__ void k_agg(const _Float16* __restrict__ s1h,
                      const float* __restrict__ s2, float* agg) {
  int b = blockIdx.x >> 3, chunk = blockIdx.x & 7;
  int lane = threadIdx.x & 31, w = threadIdx.x >> 5;
  int t = chunk * 8 + w;
  int mt = (t >> 1) * 16, nt = (t & 1) * 16;
  const _Float16* sb = s1h + (size_t)b * N_ * K1_;
  const float* s2b = s2 + (size_t)b * 64 * 32;
  __builtin_prefetch(s2b, 0, 0);  // global_prefetch_b8
  v8f c = {};
#pragma unroll
  for (int k = 0; k < 64; k += 32)
    c = wmma_f16(load_a_h(sb, 64, mt, k, lane), load_b(s2b, 32, k, nt, lane), c);
  store_c(agg + (size_t)b * N_ * 32, 32, mt, nt, lane, c);
}

// ---------------- host-side orchestration ----------------

extern "C" void kernel_launch(void* const* d_in, const int* in_sizes, int n_in,
                              void* d_out, int out_size, void* d_ws,
                              size_t ws_size, hipStream_t stream) {
  (void)in_sizes; (void)n_in; (void)out_size; (void)ws_size;
  const float* pos = (const float*)d_in[0];
  const int* edge_index = (const int*)d_in[1];
  const float* ew = (const float*)d_in[2];
  const float* W1 = (const float*)d_in[4];
  const float* b1 = (const float*)d_in[5];
  const float* Wp1 = (const float*)d_in[6];
  const float* bp1 = (const float*)d_in[7];
  const float* W2r = (const float*)d_in[8];
  const float* b2 = (const float*)d_in[9];
  const float* W2s = (const float*)d_in[10];
  const float* Wp2 = (const float*)d_in[11];
  const float* bp2 = (const float*)d_in[12];
  const float* W3r = (const float*)d_in[13];
  const float* b3 = (const float*)d_in[14];
  const float* W3s = (const float*)d_in[15];
  const int* row = edge_index;
  const int* col = edge_index + E_;

  float* out = (float*)d_out;
  float* x_out = out;              // B*32*128 = 524288
  float* mc_out = out + 524288;    // 1
  float* o_out = out + 524289;     // 1
  float* agg_out = out + 524290;   // B*512*32 = 2097152
  float* adj2n = out + 2621442;    // B*32*32 = 131072

  float* ws = (float*)d_ws;
  size_t o = 0;
  float* dis = ws + o;      o += NT_;
  float* xw = ws + o;       o += (size_t)NT_ * HID_;
  float* h = ws + o;        o += (size_t)NT_ * HID_;
  float* s1 = ws + o;       o += (size_t)NT_ * K1_;
  float* As1 = ws + o;      o += (size_t)NT_ * K1_;
  float* dflat1 = ws + o;   o += NT_;
  float* out1 = ws + o;     o += (size_t)B_ * 4096;
  float* oadj1 = ws + o;    o += (size_t)B_ * 4096;
  float* ss1 = ws + o;      o += (size_t)B_ * 4096;
  float* adj1n = ws + o;    o += (size_t)B_ * 4096;
  float* dflat2 = ws + o;   o += (size_t)B_ * 64;
  float* t1 = ws + o;       o += (size_t)B_ * 4096;
  float* x2 = ws + o;       o += (size_t)B_ * 4096;
  float* s2 = ws + o;       o += (size_t)B_ * 2048;
  float* out2 = ws + o;     o += (size_t)B_ * 2048;
  float* oadj2 = ws + o;    o += (size_t)B_ * 1024;
  float* ss2 = ws + o;      o += (size_t)B_ * 1024;
  float* t2 = ws + o;       o += (size_t)B_ * 2048;
  // fp16 mirrors for the bandwidth-bound pool-1 / agg stages
  _Float16* x1h = (_Float16*)(ws + o);  o += (size_t)NT_ * HID_ / 2;
  _Float16* s1h = (_Float16*)(ws + o);  o += (size_t)NT_ * K1_ / 2;
  _Float16* As1h = (_Float16*)(ws + o); o += (size_t)NT_ * K1_ / 2;

  // init
  k_fill<<<(NT_ + 255) / 256, 256, 0, stream>>>(dis, 1.0f, NT_);
  k_fill<<<(NT_ * K1_ + 255) / 256, 256, 0, stream>>>(As1, 0.f, NT_ * K1_);
  k_fill<<<(NT_ + 255) / 256, 256, 0, stream>>>(dflat1, 0.f, NT_);
  k_fill<<<1, 256, 0, stream>>>(mc_out, 0.f, 2);

  // GCN normalization
  k_deg_scatter<<<E_ / 256, 256, 0, stream>>>(col, ew, dis);
  k_dis<<<NT_ / 256, 256, 0, stream>>>(dis);

  // xw = pos @ W1 (WMMA)
  k_gemm_xw<<<NT_ / 32, 256, 0, stream>>>(pos, W1, xw);

  // h = segment_sum(norm * xw[row]) + b1 ; relu (+ f16 mirror)
  k_h_init<<<NT_ * HID_ / 256, 256, 0, stream>>>(dis, xw, b1, h);
  k_h_scatter<<<(int)((size_t)E_ * HID_ / 256), 256, 0, stream>>>(row, col, ew,
                                                                  dis, xw, h);
  k_relu_cvt<<<NT_ * HID_ / 256, 256, 0, stream>>>(h, x1h, NT_ * HID_);

  // s1 logits + softmax (+ f16 mirror)
  k_gemm_s1<<<NT_ / 32, 256, 0, stream>>>(h, Wp1, bp1, s1);
  k_softmax64<<<NT_ * 32 / 256, 256, 0, stream>>>(s1, s1h);

  // As1 = A @ s1 from edges (dense adjacency never materialized)
  k_As_scatter<<<(int)((size_t)E_ * K1_ / 256), 256, 0, stream>>>(row, col, s1,
                                                                  As1, dflat1);
  k_cvt_h<<<NT_ * K1_ / 256, 256, 0, stream>>>(As1, As1h, NT_ * K1_);

  // pool 1 GEMMs (TDM double-buffered) + stats/losses/normalization
  k_pool1<<<B_, 256, 0, stream>>>(s1h, x1h, As1h, out1, oadj1, ss1);
  k_stats1<<<B_, 256, 0, stream>>>(s1, dflat1, oadj1, ss1, adj1n, dflat2,
                                   mc_out, o_out);

  // dgc level 2
  k_t1<<<B_, 256, 0, stream>>>(adj1n, out1, t1);
  k_x2<<<(B_ * 64 / 16) * 4 / 8, 256, 0, stream>>>(t1, out1, W2r, W2s, b2, x2);

  // s2 logits + softmax
  k_gemm_s2<<<(B_ * 64 / 16) * 2 / 8, 256, 0, stream>>>(x2, Wp2, bp2, s2);
  k_softmax32<<<B_ * 64 * 32 / 256, 256, 0, stream>>>(s2, B_ * 64);

  // pool 2 + stats
  k_pool2<<<B_, 256, 0, stream>>>(adj1n, s2, x2, out2, oadj2, ss2);
  k_stats2<<<B_, 256, 0, stream>>>(s2, dflat2, oadj2, ss2, adj2n, mc_out, o_out);

  // dgc level 3 -> x output
  k_t2<<<B_, 256, 0, stream>>>(adj2n, out2, t2);
  k_x3<<<(B_ * 32 / 16) * 8 / 8, 256, 0, stream>>>(t2, out2, W3r, W3s, b3, x_out);

  // agg = s1s @ s2s -> agg output
  k_agg<<<B_ * 8, 256, 0, stream>>>(s1h, s2, agg_out);
}
